// GraphSAGEClassifier_79663053406465
// MI455X (gfx1250) — matrix operand order repaired and checked by reading, not compiled, not run on hardware
//
#include <hip/hip_runtime.h>

// CDNA5 / gfx1250: wave32, WMMA 16x16x32 f16 -> f32 accumulate.
typedef __attribute__((ext_vector_type(16))) _Float16 v16h;
typedef __attribute__((ext_vector_type(8)))  float    v8f;

#define DM 128   // memory_dim (K dimension of every GEMM)

// ---------------------------------------------------------------------------
// Utility: grid-stride zero
// ---------------------------------------------------------------------------
__global__ void zero_f32_kernel(float* __restrict__ p, long n) {
  long i = (long)blockIdx.x * blockDim.x + threadIdx.x;
  long stride = (long)gridDim.x * blockDim.x;
  for (; i < n; i += stride) p[i] = 0.0f;
}

// ---------------------------------------------------------------------------
// Weight prep: W [128, ncols] row-major f32  ->  out [ncols, 128] f16
// (transposed so a WMMA B-fragment per lane is 16 contiguous f16 = 32 bytes)
// ---------------------------------------------------------------------------
__global__ void convert_w_kernel(const float* __restrict__ W,
                                 _Float16* __restrict__ out, int ncols) {
  int i = blockIdx.x * blockDim.x + threadIdx.x;
  int total = DM * ncols;
  if (i >= total) return;
  int c = i / DM;
  int k = i - c * DM;
  out[i] = (_Float16)W[k * ncols + c];
}

// ---------------------------------------------------------------------------
// Edge scatter: one wave32 per edge.  agg[dst,:] += x[src,:], deg[dst] += 1.
// 128 floats/row -> float4 per lane -> 4 global_atomic_add_f32 per lane.
// Memory-bound: E*D*4B read + E*D*4B atomic RMW per layer (~53 us/call at
// 23.3 TB/s for both layers) — this dominates; GEMMs are WMMA-cheap.
// ---------------------------------------------------------------------------
__global__ void scatter_kernel(const float* __restrict__ x,
                               const int* __restrict__ src,
                               const int* __restrict__ dst,
                               float* __restrict__ agg,
                               float* __restrict__ deg,
                               int E) {
  int gid  = blockIdx.x * blockDim.x + threadIdx.x;
  int e    = gid >> 5;
  int lane = gid & 31;
  if (e >= E) return;
  int s = src[e];
  int d = dst[e];
  float4 v = ((const float4*)(x + (long)s * DM))[lane];
  float* ar = agg + (long)d * DM + lane * 4;
  atomicAdd(ar + 0, v.x);
  atomicAdd(ar + 1, v.y);
  atomicAdd(ar + 2, v.z);
  atomicAdd(ar + 3, v.w);
  if (lane == 0) atomicAdd(deg + d, 1.0f);
}

// ---------------------------------------------------------------------------
// Fused SAGE/MLP GEMM (all structural options are compile-time so the inner
// loop is branch-free: load B frag -> v_wmma -> load B frag -> v_wmma):
//   out = act( scale(A1) @ W1 + [A2 @ W2] + bias ),  K = 128 fixed
//   scale(A1) = A1 / max(deg,1) when HAS_DEG (mean aggregation)
//   rows of A1 gathered through `gather` when HAS_GATHER (classifier head)
// One wave computes a 16-row x (NCT*16)-col strip:
//   4 K-steps of 32, NCT column tiles, v_wmma_f32_16x16x32_f16 each
//   (x2 when the root term A2 is present).
// A-fragment layout (16-bit A, 16x32): lane L holds row M=L&15;
//   elems 0..7  -> K = k0 + (L>>4)*16 + i
//   elems 8..15 -> K = k0 + (L>>4)*16 + 8 + (i-8)
// B-fragment layout (16-bit B, 32x16): lane L holds col N=L&15;
//   elems 0..15 -> K = k0 + (L>>4)*16 + i  (contiguous in transposed weights)
// C/D layout: VGPR i, lane L -> row (L>>4)*8 + i, col L&15.
// No divergent control flow anywhere, so EXEC is all-ones at every WMMA.
// ---------------------------------------------------------------------------
template <int NCT, bool HAS_A2, bool HAS_DEG, bool HAS_GATHER>
__global__ void sage_wmma_kernel(const float*    __restrict__ A1,
                                 const float*    __restrict__ deg,
                                 const int*      __restrict__ gather,
                                 const _Float16* __restrict__ W1t,   // [NCT*16][128] f16
                                 const float*    __restrict__ A2,
                                 const _Float16* __restrict__ W2t,   // [NCT*16][128] f16
                                 const float*    __restrict__ bias,
                                 float*          __restrict__ out,
                                 int nrows, int do_relu) {
  const int NCOLS = NCT * 16;
  int gid  = blockIdx.x * blockDim.x + threadIdx.x;
  int wave = gid >> 5;
  int lane = threadIdx.x & 31;
  int row0 = wave * 16;
  if (row0 >= nrows) return;               // wave-uniform

  int mrow = lane & 15;
  int half = lane >> 4;

  int arow = row0 + mrow;
  if (HAS_GATHER) arow = gather[arow];
  float scale = 1.0f;
  if (HAS_DEG) scale = 1.0f / fmaxf(deg[arow], 1.0f);

  const float* a1p = A1 + (long)arow * DM;
  const float* a2p = HAS_A2 ? (A2 + (long)arow * DM) : A1;

  const v8f vzero = {0.f, 0.f, 0.f, 0.f, 0.f, 0.f, 0.f, 0.f};
  v8f acc[NCT];
#pragma unroll
  for (int ct = 0; ct < NCT; ++ct) acc[ct] = vzero;

#pragma unroll
  for (int ks = 0; ks < 4; ++ks) {
    const int k0  = ks * 32;
    const int kb0 = k0 + half * 16;
    const int kb1 = k0 + half * 16 + 8;

    v16h a1, a2;
#pragma unroll
    for (int i = 0; i < 8; ++i) {
      a1[i]     = (_Float16)(a1p[kb0 + i] * scale);
      a1[i + 8] = (_Float16)(a1p[kb1 + i] * scale);
    }
    if (HAS_A2) {
#pragma unroll
      for (int i = 0; i < 8; ++i) {
        a2[i]     = (_Float16)a2p[kb0 + i];
        a2[i + 8] = (_Float16)a2p[kb1 + i];
      }
    }

    const int kb = k0 + half * 16;         // B-fragment K base, 32B aligned
#pragma unroll
    for (int ct = 0; ct < NCT; ++ct) {
      int col = ct * 16 + mrow;
      v16h b1 = *(const v16h*)(W1t + (long)col * DM + kb);
      acc[ct] = __builtin_amdgcn_wmma_f32_16x16x32_f16(
          false, a1, false, b1, (short)0, acc[ct], false, false);
      if (HAS_A2) {
        v16h b2 = *(const v16h*)(W2t + (long)col * DM + kb);
        acc[ct] = __builtin_amdgcn_wmma_f32_16x16x32_f16(
            false, a2, false, b2, (short)0, acc[ct], false, false);
      }
    }
  }

  // Epilogue: bias + optional ReLU, f32 store
#pragma unroll
  for (int ct = 0; ct < NCT; ++ct) {
    int col = ct * 16 + mrow;
    float bv = bias[col];
#pragma unroll
    for (int i = 0; i < 8; ++i) {
      int row = row0 + half * 8 + i;
      if (row < nrows) {
        float v = acc[ct][i] + bv;
        if (do_relu) v = fmaxf(v, 0.0f);
        out[(long)row * NCOLS + col] = v;
      }
    }
  }
}

// ---------------------------------------------------------------------------
// Launch: zero -> scatter -> fused GEMM (x2 layers) -> gathered head -> logits
// ---------------------------------------------------------------------------
extern "C" void kernel_launch(void* const* d_in, const int* in_sizes, int n_in,
                              void* d_out, int out_size, void* d_ws, size_t ws_size,
                              hipStream_t stream) {
  (void)n_in; (void)out_size; (void)ws_size;

  const float* node_state = (const float*)d_in[0];
  const int*   edge_index = (const int*)d_in[1];
  const int*   label_pos  = (const int*)d_in[2];
  const float* W1l = (const float*)d_in[3];
  const float* b1l = (const float*)d_in[4];
  const float* W1r = (const float*)d_in[5];
  const float* W2l = (const float*)d_in[6];
  const float* b2l = (const float*)d_in[7];
  const float* W2r = (const float*)d_in[8];
  const float* Wc1 = (const float*)d_in[9];
  const float* bc1 = (const float*)d_in[10];
  const float* Wc2 = (const float*)d_in[11];
  const float* bc2 = (const float*)d_in[12];

  const int N = in_sizes[0] / DM;     // 100000
  const int E = in_sizes[1] / 2;      // 600000
  const int B = in_sizes[2];          // 10000
  const int C = in_sizes[12];         // 32

  const int* src = edge_index;        // edge_index[0,:]
  const int* dst = edge_index + E;    // edge_index[1,:]

  // ---- workspace layout (all fully rewritten every call) ----
  float* ws = (float*)d_ws;
  size_t o = 0;
  float* agg = ws + o; o += (size_t)N * DM;
  float* deg = ws + o; o += (size_t)N;          // contiguous after agg
  float* x1  = ws + o; o += (size_t)N * DM;
  float* x2  = ws + o; o += (size_t)N * DM;
  float* h   = ws + o; o += (size_t)B * DM;
  o = (o + 15) & ~(size_t)15;                   // 64B align for v16h loads
  _Float16* wt1l = (_Float16*)(ws + o);
  _Float16* wt1r = wt1l + DM * DM;
  _Float16* wt2l = wt1r + DM * DM;
  _Float16* wt2r = wt2l + DM * DM;
  _Float16* wtc1 = wt2r + DM * DM;
  _Float16* wtc2 = wtc1 + DM * DM;              // [C][128]

  // ---- weight convert + transpose (tiny, L2-resident afterwards) ----
  {
    int t = DM * DM, g = (t + 255) / 256;
    convert_w_kernel<<<g, 256, 0, stream>>>(W1l, wt1l, DM);
    convert_w_kernel<<<g, 256, 0, stream>>>(W1r, wt1r, DM);
    convert_w_kernel<<<g, 256, 0, stream>>>(W2l, wt2l, DM);
    convert_w_kernel<<<g, 256, 0, stream>>>(W2r, wt2r, DM);
    convert_w_kernel<<<g, 256, 0, stream>>>(Wc1, wtc1, DM);
    int t2 = DM * C, g2 = (t2 + 255) / 256;
    convert_w_kernel<<<g2, 256, 0, stream>>>(Wc2, wtc2, C);
  }

  const long zn = (long)N * DM + N;             // agg + deg together
  const int  sgrid  = (int)(((long)E * 32 + 255) / 256);
  const int  ggridN = ((N / 16) * 32 + 255) / 256;
  const int  ggridB = ((B / 16) * 32 + 255) / 256;

  // ---- layer 1 ----
  zero_f32_kernel<<<2048, 256, 0, stream>>>(agg, zn);
  scatter_kernel<<<sgrid, 256, 0, stream>>>(node_state, src, dst, agg, deg, E);
  sage_wmma_kernel<8, true, true, false><<<ggridN, 256, 0, stream>>>(
      agg, deg, nullptr, wt1l, node_state, wt1r, b1l, x1, N, 1);

  // ---- layer 2 ----
  zero_f32_kernel<<<2048, 256, 0, stream>>>(agg, zn);
  scatter_kernel<<<sgrid, 256, 0, stream>>>(x1, src, dst, agg, deg, E);
  sage_wmma_kernel<8, true, true, false><<<ggridN, 256, 0, stream>>>(
      agg, deg, nullptr, wt2l, x1, wt2r, b2l, x2, N, 1);

  // ---- classifier head: gather + Linear(128,128)+ReLU, then Linear(128,32) ----
  sage_wmma_kernel<8, false, false, true><<<ggridB, 256, 0, stream>>>(
      x2, nullptr, label_pos, wtc1, nullptr, nullptr, bc1, h, B, 1);
  sage_wmma_kernel<2, false, false, false><<<ggridB, 256, 0, stream>>>(
      h, nullptr, nullptr, wtc2, nullptr, nullptr, bc2, (float*)d_out, B, 0);
}
